// CLAM_Instance_3427383902752
// MI455X (gfx1250) — compile-verified, matrix operand order checked by reading, not run on hardware
//
#include <hip/hip_runtime.h>
#include <hip/hip_bf16.h>
#include <float.h>

// ---------------------------------------------------------------------------
// Problem constants: N=50000, D_IN=1024, D_H=512, D_A=256, N_CLS=2, K=8.
// 50000 = 3125 * 16 -> exact 16-row M-tiles.
// ---------------------------------------------------------------------------
#define N_INST 50000
#define D_IN   1024
#define D_H    512
#define D_A    256
#define KSEL   8
#define M_TILES (N_INST / 16)   // 3125

typedef __attribute__((ext_vector_type(16))) __bf16 v16bf;
typedef __attribute__((ext_vector_type(8)))  __bf16 v8bf;
typedef __attribute__((ext_vector_type(8)))  float  v8f;

// LDS tile row stride in bf16 elements: 32 k-values + 8 pad.
// 80B stride: 16B aligned (ds_load_b128) and bank-conflict-benign.
#define BSTRIDE 40

// ---------------------------------------------------------------------------
// CDNA5 async global->LDS copy (ASYNCcnt tracked, no VGPR transit).
// vdst = LDS byte offset (VGPR), vaddr = 64-bit global address.
// ---------------------------------------------------------------------------
__device__ __forceinline__ void async_load_b128(unsigned lds_off, const void* gaddr) {
  asm volatile("global_load_async_to_lds_b128 %0, %1, off"
               :: "v"(lds_off), "v"(gaddr) : "memory");
}
__device__ __forceinline__ void wait_async_all() {
  asm volatile("s_wait_asynccnt 0" ::: "memory");
}
__device__ __forceinline__ unsigned lds_off_of(const void* p) {
  return (unsigned)(unsigned long long)p;   // low 32 bits of LDS flat addr = LDS offset
}

// A/B fragment for v_wmma_f32_16x16x32_bf16: two contiguous 16B runs
// (k0..k0+7 and k0+16..k0+23) from a row-major-over-K row.
__device__ __forceinline__ v16bf load_bfrag(const __bf16* p) {
  v8bf lo = *(const v8bf*)(p);
  v8bf hi = *(const v8bf*)(p + 16);
  v16bf r;
#pragma unroll
  for (int i = 0; i < 8; ++i) { r[i] = lo[i]; r[i + 8] = hi[i]; }
  return r;
}

__device__ __forceinline__ v8f wmma_bf16(v16bf a, v16bf b, v8f c) {
  return __builtin_amdgcn_wmma_f32_16x16x32_bf16(false, a, false, b,
                                                 (short)0, c, false, false);
}

// ---------------------------------------------------------------------------
// K0: convert fc_W (512x1024), att_a_W (256x512), att_b_W (256x512) to bf16.
// ---------------------------------------------------------------------------
__global__ void __launch_bounds__(1024)
convert_weights(const float* __restrict__ fcW, const float* __restrict__ Wa,
                const float* __restrict__ Wb, __bf16* __restrict__ fcWb,
                __bf16* __restrict__ WaB, __bf16* __restrict__ WbB) {
  const int total = D_H * D_IN + 2 * D_A * D_H;
  for (int i = blockIdx.x * blockDim.x + threadIdx.x; i < total;
       i += gridDim.x * blockDim.x) {
    if (i < D_H * D_IN) {
      fcWb[i] = (__bf16)fcW[i];
    } else if (i < D_H * D_IN + D_A * D_H) {
      int j = i - D_H * D_IN;
      WaB[j] = (__bf16)Wa[j];
    } else {
      int j = i - D_H * D_IN - D_A * D_H;
      WbB[j] = (__bf16)Wb[j];
    }
  }
}

// ---------------------------------------------------------------------------
// K1: hh = relu(x @ fc_W^T + fc_b), stored bf16.
// Block = 8 waves = 8 M-tiles (128 rows). N=512 in 2 chunks of 256 cols
// (16 acc tiles = 128 VGPRs per wave).  Per k-step the 256x32 fc_W slice
// (16KB) is staged into LDS with async-to-LDS loads, double buffered, and
// shared by all 8 waves.  x (fp32) is converted to bf16 in-register with a
// one-step software prefetch.
// ---------------------------------------------------------------------------
__global__ void __launch_bounds__(256)
gemm_hh(const float* __restrict__ x, const __bf16* __restrict__ fcWb,
        const float* __restrict__ fcb, __bf16* __restrict__ hhB) {
  __shared__ __align__(16) __bf16 Bsm[2][256 * BSTRIDE];   // 2 x 20KB
  const int tid  = threadIdx.x;
  const int wave = tid >> 5;
  const int lane = tid & 31;
  int tile = blockIdx.x * 8 + wave;
  if (tile >= M_TILES) tile = M_TILES - 1;   // clamp: keep all waves in barriers
  const int half = lane >> 4;
  const int l15  = lane & 15;
  const float* xrow = x + (size_t)(tile * 16 + l15) * D_IN;

  // staging map: 4 consecutive threads fetch one 64B k-row (parts of 16B)
  const int part = tid & 3;
  const int nl0  = tid >> 2;   // 0..63; 4 iterations cover 256 rows

  for (int chunk = 0; chunk < 2; ++chunk) {
    const int nbase = chunk * 256;
    // prologue: stage ks=0 into buffer 0
#pragma unroll
    for (int it = 0; it < 4; ++it) {
      const int nl = nl0 + it * 64;
      async_load_b128(lds_off_of(&Bsm[0][nl * BSTRIDE + part * 8]),
                      fcWb + (size_t)(nbase + nl) * D_IN + part * 8);
    }
    wait_async_all();
    __syncthreads();

    v8f acc[16] = {};
    v8f flo = *(const v8f*)(xrow + half * 8);
    v8f fhi = *(const v8f*)(xrow + half * 8 + 16);

    for (int ks = 0; ks < D_IN / 32; ++ks) {
      const int cur = ks & 1;
      if (ks + 1 < D_IN / 32) {   // stage next k-slice while computing
#pragma unroll
        for (int it = 0; it < 4; ++it) {
          const int nl = nl0 + it * 64;
          async_load_b128(lds_off_of(&Bsm[cur ^ 1][nl * BSTRIDE + part * 8]),
                          fcWb + (size_t)(nbase + nl) * D_IN + (ks + 1) * 32 + part * 8);
        }
      }
      v16bf afrag;
#pragma unroll
      for (int i = 0; i < 8; ++i) {
        afrag[i]     = (__bf16)flo[i];
        afrag[i + 8] = (__bf16)fhi[i];
      }
      if (ks + 1 < D_IN / 32) {   // prefetch next A slice
        const int k0n = (ks + 1) * 32 + half * 8;
        flo = *(const v8f*)(xrow + k0n);
        fhi = *(const v8f*)(xrow + k0n + 16);
      }
#pragma unroll
      for (int t = 0; t < 16; ++t) {
        const __bf16* bp = &Bsm[cur][(t * 16 + l15) * BSTRIDE + half * 8];
        acc[t] = wmma_bf16(afrag, load_bfrag(bp), acc[t]);
      }
      wait_async_all();
      __syncthreads();
    }
    // epilogue: bias + relu + bf16 store (lane holds col n, rows m = 8*half+r)
#pragma unroll
    for (int t = 0; t < 16; ++t) {
      const int n = nbase + t * 16 + l15;
      const float bias = fcb[n];
#pragma unroll
      for (int r = 0; r < 8; ++r) {
        const int m = tile * 16 + half * 8 + r;
        float v = acc[t][r] + bias;
        v = v > 0.f ? v : 0.f;
        hhB[(size_t)m * D_H + n] = (__bf16)v;
      }
    }
  }
}

// ---------------------------------------------------------------------------
// K2: instance_scores = hh @ cls_W^T + cls_b (N x 2).  Wave per row, each
// lane reads its 16-element slice with two b128 loads; hh is L2-resident.
// ---------------------------------------------------------------------------
__global__ void __launch_bounds__(256)
inst_scores(const __bf16* __restrict__ hhB, const float* __restrict__ clsW,
            const float* __restrict__ clsb, float* __restrict__ out_sc) {
  const int row  = blockIdx.x * 8 + (threadIdx.x >> 5);
  const int lane = threadIdx.x & 31;
  if (row >= N_INST) return;
  const __bf16* hr = hhB + (size_t)row * D_H + lane * 16;
  v8bf h0 = *(const v8bf*)(hr);
  v8bf h1 = *(const v8bf*)(hr + 8);
  const float* c0 = clsW + lane * 16;
  const float* c1 = clsW + D_H + lane * 16;
  v8f w00 = *(const v8f*)c0, w01 = *(const v8f*)(c0 + 8);
  v8f w10 = *(const v8f*)c1, w11 = *(const v8f*)(c1 + 8);
  float s0 = 0.f, s1 = 0.f;
#pragma unroll
  for (int i = 0; i < 8; ++i) {
    s0 += (float)h0[i] * w00[i] + (float)h1[i] * w01[i];
    s1 += (float)h0[i] * w10[i] + (float)h1[i] * w11[i];
  }
#pragma unroll
  for (int m = 16; m >= 1; m >>= 1) {
    s0 += __shfl_xor(s0, m, 32);
    s1 += __shfl_xor(s1, m, 32);
  }
  if (lane == 0) {
    out_sc[row * 2 + 0] = s0 + clsb[0];
    out_sc[row * 2 + 1] = s1 + clsb[1];
  }
}

// ---------------------------------------------------------------------------
// K3: gated attention, fused:
//   A_raw[m] = sum_n tanh(hh@Wa^T+ba)[m,n] * sigmoid(hh@Wb^T+bb)[m,n] * wc[n] + cb
// Same async-LDS double-buffer scheme; Wa rows 0..63 and Wb rows 64..127
// share one 128-row k-slice tile.  D_A=256 in 4 chunks of 64 cols.
// ---------------------------------------------------------------------------
__global__ void __launch_bounds__(256)
attention(const __bf16* __restrict__ hhB, const __bf16* __restrict__ WaB,
          const float* __restrict__ ab, const __bf16* __restrict__ WbB,
          const float* __restrict__ bb, const float* __restrict__ wc,
          const float* __restrict__ cb, float* __restrict__ A_raw) {
  __shared__ __align__(16) __bf16 Bsm[2][128 * BSTRIDE];   // 2 x 10KB
  const int tid  = threadIdx.x;
  const int wave = tid >> 5;
  const int lane = tid & 31;
  int tile = blockIdx.x * 8 + wave;
  if (tile >= M_TILES) tile = M_TILES - 1;
  const int half = lane >> 4;
  const int l15  = lane & 15;
  const __bf16* hrow = hhB + (size_t)(tile * 16 + l15) * D_H;

  const int part = tid & 3;
  const int nl0  = tid >> 2;   // 0..63; it=0 -> Wa rows, it=1 -> Wb rows

  float rp[8] = {};
  for (int chunk = 0; chunk < 4; ++chunk) {
    const int nbase = chunk * 64;
    // prologue: stage ks=0 into buffer 0
#pragma unroll
    for (int it = 0; it < 2; ++it) {
      const int nl = nl0 + it * 64;
      const __bf16* src = (it == 0 ? WaB + (size_t)(nbase + nl0) * D_H
                                   : WbB + (size_t)(nbase + nl0) * D_H) + part * 8;
      async_load_b128(lds_off_of(&Bsm[0][nl * BSTRIDE + part * 8]), src);
    }
    wait_async_all();
    __syncthreads();

    v8f accA[4] = {};
    v8f accB[4] = {};
    for (int ks = 0; ks < D_H / 32; ++ks) {
      const int cur = ks & 1;
      if (ks + 1 < D_H / 32) {
#pragma unroll
        for (int it = 0; it < 2; ++it) {
          const int nl = nl0 + it * 64;
          const __bf16* src = (it == 0 ? WaB + (size_t)(nbase + nl0) * D_H
                                       : WbB + (size_t)(nbase + nl0) * D_H)
                              + (ks + 1) * 32 + part * 8;
          async_load_b128(lds_off_of(&Bsm[cur ^ 1][nl * BSTRIDE + part * 8]), src);
        }
      }
      const v16bf afrag = load_bfrag(hrow + ks * 32 + half * 8);
#pragma unroll
      for (int t = 0; t < 4; ++t) {
        const __bf16* pa = &Bsm[cur][(t * 16 + l15) * BSTRIDE + half * 8];
        const __bf16* pb = &Bsm[cur][(64 + t * 16 + l15) * BSTRIDE + half * 8];
        accA[t] = wmma_bf16(afrag, load_bfrag(pa), accA[t]);
        accB[t] = wmma_bf16(afrag, load_bfrag(pb), accB[t]);
      }
      wait_async_all();
      __syncthreads();
    }
#pragma unroll
    for (int t = 0; t < 4; ++t) {
      const int n = nbase + t * 16 + l15;
      const float biasa = ab[n], biasb = bb[n], w = wc[n];
#pragma unroll
      for (int r = 0; r < 8; ++r) {
        const float av = tanhf(accA[t][r] + biasa);
        const float bv = 1.f / (1.f + __expf(-(accB[t][r] + biasb)));
        rp[r] += av * bv * w;
      }
    }
  }
  // lanes 0..15 hold rows 0..7 (one column each); lanes 16..31 rows 8..15.
#pragma unroll
  for (int r = 0; r < 8; ++r) {
#pragma unroll
    for (int m = 8; m >= 1; m >>= 1) rp[r] += __shfl_xor(rp[r], m, 32);
  }
  if (l15 == 0) {
    const float c = cb[0];
#pragma unroll
    for (int r = 0; r < 8; ++r)
      A_raw[tile * 16 + half * 8 + r] = rp[r] + c;
  }
}

// ---------------------------------------------------------------------------
// K4: preds[c] = sum_n sc[n,c]*e^{r_n-max} / sum_n e^{r_n-max}
// ---------------------------------------------------------------------------
__global__ void __launch_bounds__(1024)
preds_kernel(const float* __restrict__ A_raw, const float* __restrict__ sc,
             float* __restrict__ out) {
  __shared__ float red[1024];
  const int tid = threadIdx.x;
  float m = -FLT_MAX;
  for (int n = tid; n < N_INST; n += 1024) m = fmaxf(m, A_raw[n]);
  red[tid] = m;
  __syncthreads();
  for (int s = 512; s > 0; s >>= 1) {
    if (tid < s) red[tid] = fmaxf(red[tid], red[tid + s]);
    __syncthreads();
  }
  const float M = red[0];
  __syncthreads();

  float den = 0.f, n0 = 0.f, n1 = 0.f;
  for (int n = tid; n < N_INST; n += 1024) {
    const float e = __expf(A_raw[n] - M);
    den += e;
    n0 += sc[2 * n + 0] * e;
    n1 += sc[2 * n + 1] * e;
  }
  float res[3];
  float in[3] = {den, n0, n1};
#pragma unroll
  for (int k = 0; k < 3; ++k) {
    red[tid] = in[k];
    __syncthreads();
    for (int s = 512; s > 0; s >>= 1) {
      if (tid < s) red[tid] += red[tid + s];
      __syncthreads();
    }
    res[k] = red[0];
    __syncthreads();
  }
  if (tid == 0) {
    out[0] = res[1] / res[0];
    out[1] = res[2] / res[0];
  }
}

// ---------------------------------------------------------------------------
// K5: iterative top-8 / bottom-8 selection, then 16x2 instance logits and
// mean cross-entropy loss.
// ---------------------------------------------------------------------------
__global__ void __launch_bounds__(1024)
topk_loss(const float* __restrict__ A_raw, float* __restrict__ scratch,
          const __bf16* __restrict__ hhB, const float* __restrict__ instW,
          const float* __restrict__ instb, float* __restrict__ out_loss) {
  __shared__ float rv[1024];
  __shared__ int   ri[1024];
  __shared__ int   ids[16];
  __shared__ float logits[32];
  const int tid = threadIdx.x;

  for (int phase = 0; phase < 2; ++phase) {
    for (int n = tid; n < N_INST; n += 1024)
      scratch[n] = phase ? -A_raw[n] : A_raw[n];
    __syncthreads();
    for (int s = 0; s < KSEL; ++s) {
      float bv = -FLT_MAX;
      int bi = 0x7fffffff;
      for (int n = tid; n < N_INST; n += 1024) {
        const float v = scratch[n];
        if (v > bv || (v == bv && n < bi)) { bv = v; bi = n; }
      }
      rv[tid] = bv; ri[tid] = bi;
      __syncthreads();
      for (int st = 512; st > 0; st >>= 1) {
        if (tid < st) {
          if (rv[tid + st] > rv[tid] ||
              (rv[tid + st] == rv[tid] && ri[tid + st] < ri[tid])) {
            rv[tid] = rv[tid + st];
            ri[tid] = ri[tid + st];
          }
        }
        __syncthreads();
      }
      if (tid == 0) {
        ids[phase * KSEL + s] = ri[0];
        scratch[ri[0]] = -FLT_MAX;
      }
      __syncthreads();
    }
  }

  const int wv = tid >> 5, lane = tid & 31;
  if (wv < 16) {
    const __bf16* hr = hhB + (size_t)ids[wv] * D_H;
    float s0 = 0.f, s1 = 0.f;
    for (int j = lane; j < D_H; j += 32) {
      const float hv = (float)hr[j];
      s0 += hv * instW[j];
      s1 += hv * instW[D_H + j];
    }
#pragma unroll
    for (int m = 16; m >= 1; m >>= 1) {
      s0 += __shfl_xor(s0, m, 32);
      s1 += __shfl_xor(s1, m, 32);
    }
    if (lane == 0) {
      logits[wv * 2 + 0] = s0 + instb[0];
      logits[wv * 2 + 1] = s1 + instb[1];
    }
  }
  __syncthreads();
  if (tid == 0) {
    float loss = 0.f;
    for (int i = 0; i < 16; ++i) {
      const float l0 = logits[2 * i], l1 = logits[2 * i + 1];
      const float mx = fmaxf(l0, l1);
      const float lse = mx + __logf(__expf(l0 - mx) + __expf(l1 - mx));
      const float lp = ((i < KSEL) ? l1 : l0) - lse;
      loss -= lp;
    }
    out_loss[0] = loss / 16.f;
  }
}

// ---------------------------------------------------------------------------
// Launch.  Output layout: [0:2] preds, [2:100002] instance_scores,
// [100002:150002] A_raw, [150002] loss.
// Workspace: bf16 weights (~1.5MB) + bf16 hh (51.2MB) + 200KB selection scratch.
// ---------------------------------------------------------------------------
extern "C" void kernel_launch(void* const* d_in, const int* in_sizes, int n_in,
                              void* d_out, int out_size, void* d_ws, size_t ws_size,
                              hipStream_t stream) {
  (void)in_sizes; (void)n_in; (void)out_size; (void)ws_size;
  const float* h     = (const float*)d_in[0];
  const float* fcW   = (const float*)d_in[1];
  const float* fcb   = (const float*)d_in[2];
  const float* aW    = (const float*)d_in[3];
  const float* ab    = (const float*)d_in[4];
  const float* bW    = (const float*)d_in[5];
  const float* bb    = (const float*)d_in[6];
  const float* cW    = (const float*)d_in[7];
  const float* cb    = (const float*)d_in[8];
  const float* clsW  = (const float*)d_in[9];
  const float* clsb  = (const float*)d_in[10];
  const float* instW = (const float*)d_in[11];
  const float* instb = (const float*)d_in[12];
  // d_in[13] = label (unused by the reference computation)

  float* out     = (float*)d_out;
  float* preds   = out;
  float* inst_sc = out + 2;
  float* A_raw   = out + 2 + N_INST * 2;
  float* loss    = out + 2 + N_INST * 2 + N_INST;

  __bf16* fcWb = (__bf16*)d_ws;
  __bf16* WaB  = fcWb + (size_t)D_H * D_IN;
  __bf16* WbB  = WaB + (size_t)D_A * D_H;
  __bf16* hhB  = WbB + (size_t)D_A * D_H;
  float* scratch = (float*)(hhB + (size_t)N_INST * D_H);

  hipLaunchKernelGGL(convert_weights, dim3(768), dim3(1024), 0, stream,
                     fcW, aW, bW, fcWb, WaB, WbB);
  hipLaunchKernelGGL(gemm_hh, dim3((M_TILES + 7) / 8), dim3(256), 0, stream,
                     h, fcWb, fcb, hhB);
  hipLaunchKernelGGL(inst_scores, dim3(N_INST / 8), dim3(256), 0, stream,
                     hhB, clsW, clsb, inst_sc);
  hipLaunchKernelGGL(attention, dim3((M_TILES + 7) / 8), dim3(256), 0, stream,
                     hhB, WaB, ab, WbB, bb, cW, cb, A_raw);
  hipLaunchKernelGGL(preds_kernel, dim3(1), dim3(1024), 0, stream,
                     A_raw, inst_sc, preds);
  hipLaunchKernelGGL(topk_loss, dim3(1), dim3(1024), 0, stream,
                     A_raw, scratch, hhB, instW, instb, loss);
}
